// DVSNet_84215718740085
// MI455X (gfx1250) — matrix operand order repaired and checked by reading, not compiled, not run on hardware
//
#include <hip/hip_runtime.h>

typedef _Float16 h16;
typedef __attribute__((ext_vector_type(16))) _Float16 v16h;
typedef __attribute__((ext_vector_type(8)))  float    v8f;
typedef unsigned int u32;

union Frag { v16h v; uint4 q[2]; };

// WAGE quantizer, bits=8: delta = 2^-7; round-nearest-even (matches jnp.round),
// clamp to [-1+delta, 1-delta] = [-127/128, 127/128].
__device__ __forceinline__ float wage_q8(float v) {
  float q = rintf(v * 128.0f);
  q = fmaxf(fminf(q, 127.0f), -127.0f);
  return q * 0.0078125f;
}

template<int H> struct Log2;
template<> struct Log2<64> { static constexpr int v = 6; };
template<> struct Log2<32> { static constexpr int v = 5; };
template<> struct Log2<16> { static constexpr int v = 4; };
template<> struct Log2<8>  { static constexpr int v = 3; };
template<> struct Log2<4>  { static constexpr int v = 2; };

// ---------------------------------------------------------------------------
// Generic 3x3 conv, stride 1, pad 1, Cin=Cout=32.
// Activations NHWC f16 with a physical 1-pixel ZERO HALO: [(H+2)x(W+2)x32].
// One wave per (16 consecutive logical pixels) x (32 oc) tile.
// K=32 == Cin: 9 taps x 2 oc-tiles = 18 x v_wmma_f32_16x16x32_f16 per wave.
// Halo makes all A-fragment loads unconditional: 2 x global_load_b128 per tap
// with compile-time immediate offsets (H is a template parameter).
// B fragments pre-permuted to ISA per-lane layout, staged in LDS.
// ---------------------------------------------------------------------------
template<int H>
__global__ void __launch_bounds__(256)
conv3x3_wmma_k(const h16* __restrict__ in, h16* __restrict__ out,
               const h16* __restrict__ wfrag)
{
  constexpr int W      = H;
  constexpr int Wp     = W + 2;
  constexpr int P2     = Wp * Wp;
  constexpr int shiftW = Log2<H>::v;
  constexpr int log2P  = 2 * shiftW - 4;      // tiles per image = H*W/16

  __shared__ __align__(16) u32 lds_w[4608];   // 18 KB: 9*2*32*16 halfs
  const u32* wsrc = reinterpret_cast<const u32*>(wfrag);
  for (int i = threadIdx.x; i < 4608; i += 256) lds_w[i] = wsrc[i];
  __syncthreads();

  const int lane = threadIdx.x & 31;
  const int wave = threadIdx.x >> 5;
  const unsigned tile = blockIdx.x * 8u + wave;          // grid sized exactly
  const int b       = tile >> log2P;
  const int pixbase = (tile & ((1u << log2P) - 1u)) << 4;
  const int m   = lane & 15;
  const int pix = pixbase + m;
  const int y   = pix >> shiftW;
  const int x   = pix & (W - 1);
  const int hi  = (lane >= 16) ? 8 : 0;                  // K-half per A layout

  // Per-lane base at padded (y+1, x+1); all 9 taps are immediate offsets.
  const h16* base = in + ((size_t)b * P2 + (size_t)(y + 1) * Wp + (x + 1)) * 32 + hi;

  v8f c0 = {};
  v8f c1 = {};

#pragma unroll
  for (int tap = 0; tap < 9; ++tap) {
    constexpr int dummy = 0; (void)dummy;
    const int ty = tap / 3 - 1, tx = tap % 3 - 1;

    // A fragment: lane<16 -> row M=lane, halfs {K0..7,K16..23};
    //             lane>=16 -> {K8..15,K24..31}. Halo guarantees validity.
    Frag a;
    const h16* p = base + (ty * Wp + tx) * 32;
    a.q[0] = *reinterpret_cast<const uint4*>(p);         // global_load_b128
    a.q[1] = *reinterpret_cast<const uint4*>(p + 16);

    // B fragments from LDS (pre-permuted per-lane layout).
    Frag b0, b1;
    const char* l0 = reinterpret_cast<const char*>(lds_w) + ((size_t)((tap * 2 + 0) * 32 + lane)) * 32;
    const char* l1 = reinterpret_cast<const char*>(lds_w) + ((size_t)((tap * 2 + 1) * 32 + lane)) * 32;
    b0.q[0] = *reinterpret_cast<const uint4*>(l0);       // ds_load_b128
    b0.q[1] = *reinterpret_cast<const uint4*>(l0 + 16);
    b1.q[0] = *reinterpret_cast<const uint4*>(l1);
    b1.q[1] = *reinterpret_cast<const uint4*>(l1 + 16);

    c0 = __builtin_amdgcn_wmma_f32_16x16x32_f16(false, a.v, false, b0.v, (short)0, c0, false, false);
    c1 = __builtin_amdgcn_wmma_f32_16x16x32_f16(false, a.v, false, b1.v, (short)0, c1, false, false);
  }

  // D layout: VGPR r, lanes 0-15 -> (M=r, N=lane); lanes 16-31 -> (M=8+r, N=lane-16)
  const int n     = lane & 15;
  const int mbase = (lane >= 16) ? 8 : 0;
#pragma unroll
  for (int r = 0; r < 8; ++r) {
    const int opix = pixbase + mbase + r;
    const int oy = opix >> shiftW, ox = opix & (W - 1);
    h16* op = out + ((size_t)b * P2 + (size_t)(oy + 1) * Wp + (ox + 1)) * 32;
    op[n]      = (h16)wage_q8(c0[r]);
    op[16 + n] = (h16)wage_q8(c1[r]);
  }
}

// Zero the 1-pixel halo ring of a padded NHWC buffer (all 32 channels).
__global__ void __launch_bounds__(256)
halo_zero_k(h16* __restrict__ buf, int Wp, int P2, unsigned total)
{
  const unsigned idx = blockIdx.x * 256u + threadIdx.x;  // over 512*P2 pixels
  if (idx >= total) return;
  const int p = idx % P2;
  const int y = p / Wp, x = p % Wp;
  if (y == 0 || y == Wp - 1 || x == 0 || x == Wp - 1) {
    uint4 z = {0, 0, 0, 0};
    uint4* o = reinterpret_cast<uint4*>(buf + (size_t)idx * 32);
    o[0] = z; o[1] = z; o[2] = z; o[3] = z;
  }
}

// out[b,oy,ox,c] = max over 2x2 of (1 - x1)*x   (padded NHWC f16 in and out)
template<int H>
__global__ void __launch_bounds__(256)
gate_pool_k(const h16* __restrict__ x, const h16* __restrict__ x1,
            h16* __restrict__ out)
{
  constexpr int W = H, Wp = W + 2, P2 = Wp * Wp;
  constexpr int Hh = H / 2, Wh = W / 2, Wph = Wh + 2, P2h = Wph * Wph;
  constexpr int sh = Log2<H>::v - 1;

  const unsigned idx = blockIdx.x * 256u + threadIdx.x;  // logical (b,oy,ox,c)
  const int c = idx & 31;
  unsigned r = idx >> 5;
  const int ox = r & (Wh - 1); r >>= sh;
  const int oy = r & (Hh - 1); r >>= sh;
  const int b = r;

  const size_t i00 = ((size_t)b * P2 + (size_t)(2 * oy + 1) * Wp + (2 * ox + 1)) * 32 + c;
  const size_t dy  = (size_t)Wp * 32;
  const float g00 = (1.0f - (float)x1[i00])           * (float)x[i00];
  const float g01 = (1.0f - (float)x1[i00 + 32])      * (float)x[i00 + 32];
  const float g10 = (1.0f - (float)x1[i00 + dy])      * (float)x[i00 + dy];
  const float g11 = (1.0f - (float)x1[i00 + dy + 32]) * (float)x[i00 + dy + 32];
  out[((size_t)b * P2h + (size_t)(oy + 1) * Wph + (ox + 1)) * 32 + c] =
      (h16)fmaxf(fmaxf(g00, g01), fmaxf(g10, g11));
}

// [512,2,64,64] f32 NCHW -> padded [512,66,66,32] f16 NHWC interior,
// channels 2..31 zeroed.
__global__ void __launch_bounds__(256)
prep_input_k(const float* __restrict__ xin, h16* __restrict__ out)
{
  const unsigned idx = blockIdx.x * 256u + threadIdx.x;  // 512*4096
  const unsigned b = idx >> 12, p = idx & 4095u;
  const unsigned y = p >> 6, x = p & 63u;
  union { h16 h[32]; uint4 q[4]; } u;
#pragma unroll
  for (int i = 0; i < 32; ++i) u.h[i] = (h16)0.0f;
  u.h[0] = (h16)xin[(size_t)b * 8192 + p];
  u.h[1] = (h16)xin[(size_t)b * 8192 + 4096 + p];
  uint4* o = reinterpret_cast<uint4*>(out + ((size_t)b * 4356 + (size_t)(y + 1) * 66 + (x + 1)) * 32);
  o[0] = u.q[0]; o[1] = u.q[1]; o[2] = u.q[2]; o[3] = u.q[3];
}

// Quantize conv weights and permute into WMMA B-fragment layout:
// dst[tap][octile][lane][j] = Q(w[oc][ic][tap]), oc = octile*16 + lane%16,
// ic = (j<8 ? j : j+8) + (lane<16 ? 0 : 8); zero-padded beyond Cin (conv1: Cin=2).
__global__ void __launch_bounds__(256)
prep_wconv_k(const float* __restrict__ w, h16* __restrict__ dst, int Cin)
{
  const int idx  = blockIdx.x * 256 + threadIdx.x;       // 9216 exact
  const int j    = idx & 15;
  const int lane = (idx >> 4) & 31;
  const int t    = (idx >> 9) & 1;
  const int tap  = idx >> 10;
  const int oc   = t * 16 + (lane & 15);
  const int ic   = ((j < 8) ? j : j + 8) + ((lane < 16) ? 0 : 8);
  const float v  = (ic < Cin) ? w[(size_t)(oc * Cin + ic) * 9 + tap] : 0.0f;
  dst[idx] = (h16)wage_q8(v);
}

__global__ void __launch_bounds__(256)
prep_wfc_k(const float* __restrict__ w, h16* __restrict__ dst)
{
  const int idx = blockIdx.x * 256 + threadIdx.x;
  if (idx < 1408) dst[idx] = (h16)wage_q8(w[idx]);
}

// FC: act is padded NHWC [512,4,4,32] (H=2 interior); reference flattens NCHW
// (c,h,w): k = c*4 + h*2 + w.
__global__ void __launch_bounds__(256)
fc_k(const h16* __restrict__ act, const h16* __restrict__ wq, float* __restrict__ out)
{
  const int idx = blockIdx.x * 256 + threadIdx.x;        // 5632 exact
  const int b = idx / 11, o = idx % 11;
  float acc = 0.0f;
#pragma unroll 4
  for (int k = 0; k < 128; ++k) {
    const int c = k >> 2, hh = (k >> 1) & 1, ww = k & 1;
    const float a = (float)act[((size_t)b * 16 + (size_t)(hh + 1) * 4 + (ww + 1)) * 32 + c];
    acc += a * (float)wq[o * 128 + k];
  }
  out[idx] = wage_q8(acc);
}

// ---------------------------------------------------------------------------

template<int Hs>
static void run_stage(h16* const* slot, int& xs, int& ts, int& qs, h16* wfrag,
                      int widx, hipStream_t stream)
{
  constexpr int Wp = Hs + 2, P2 = Wp * Wp;
  const unsigned total = 512u * P2;
  const int hzb = (int)((total + 255u) / 256u);
  const int cblocks = 64 * ((Hs * Hs) / 16);             // 512*P/8 waves
  halo_zero_k<<<hzb, 256, 0, stream>>>(slot[xs], Wp, P2, total);
  conv3x3_wmma_k<Hs><<<cblocks, 256, 0, stream>>>(slot[xs], slot[ts],
                                                  wfrag + (size_t)widx * 9216);
  halo_zero_k<<<hzb, 256, 0, stream>>>(slot[ts], Wp, P2, total);
  conv3x3_wmma_k<Hs><<<cblocks, 256, 0, stream>>>(slot[ts], slot[qs],
                                                  wfrag + (size_t)(widx + 1) * 9216);
  const int pblocks = 64 * ((Hs / 2) * (Hs / 2));
  gate_pool_k<Hs><<<pblocks, 256, 0, stream>>>(slot[xs], slot[qs], slot[ts]);
  const int tmp = xs; xs = ts; ts = tmp;
}

extern "C" void kernel_launch(void* const* d_in, const int* in_sizes, int n_in,
                              void* d_out, int out_size, void* d_ws, size_t ws_size,
                              hipStream_t stream) {
  (void)in_sizes; (void)n_in; (void)out_size; (void)ws_size;

  // Workspace layout:
  //   [0, 202752)            : 11 conv-weight fragment blocks (9216 halfs each)
  //   [204800, 207616)       : quantized FC weights (1408 halfs)
  //   [1MB, 1MB + 3*~136MB)  : three rotating padded NHWC f16 activation slots
  h16* wfrag = (h16*)d_ws;
  h16* wfcq  = (h16*)((char*)d_ws + 204800);
  char* abase = (char*)d_ws + (1u << 20);
  const size_t SLOT = 512ull * 66 * 66 * 32 * 2;         // 142,737,408 B
  h16* slot[3] = { (h16*)abase, (h16*)(abase + SLOT), (h16*)(abase + 2 * SLOT) };

  // Weight prep: w1 (Cin=2, zero-padded), w2..w11 (Cin=32), wfc.
  prep_wconv_k<<<36, 256, 0, stream>>>((const float*)d_in[1], wfrag, 2);
  for (int i = 2; i <= 11; ++i)
    prep_wconv_k<<<36, 256, 0, stream>>>((const float*)d_in[i],
                                         wfrag + (size_t)(i - 1) * 9216, 32);
  prep_wfc_k<<<6, 256, 0, stream>>>((const float*)d_in[12], wfcq);

  // Input -> padded NHWC f16 (channel-padded to 32), then ring-zero + conv1.
  prep_input_k<<<8192, 256, 0, stream>>>((const float*)d_in[0], slot[1]);
  {
    const unsigned total = 512u * 4356u;
    halo_zero_k<<<(int)((total + 255u) / 256u), 256, 0, stream>>>(slot[1], 66, 4356, total);
  }
  conv3x3_wmma_k<64><<<16384, 256, 0, stream>>>(slot[1], slot[0], wfrag);

  int xs = 0, ts = 1, qs = 2;
  run_stage<64>(slot, xs, ts, qs, wfrag, 1, stream);
  run_stage<32>(slot, xs, ts, qs, wfrag, 3, stream);
  run_stage<16>(slot, xs, ts, qs, wfrag, 5, stream);
  run_stage<8 >(slot, xs, ts, qs, wfrag, 7, stream);
  run_stage<4 >(slot, xs, ts, qs, wfrag, 9, stream);

  fc_k<<<22, 256, 0, stream>>>(slot[xs], wfcq, (float*)d_out);
}
